// ChatBot_9517647528131
// MI455X (gfx1250) — matrix operand order repaired and checked by reading, not compiled, not run on hardware
//
#include <hip/hip_runtime.h>
#include <hip/hip_bf16.h>

// ---------------------------------------------------------------------------
// ChatBot seq2seq (encoder LSTM -> decoder LSTM -> vocab projection), CDNA5.
// GEMMs on v_wmma_f32_16x16x32_bf16; big GEMMs stage A-tiles into LDS with
// global_load_async_to_lds_b128 (ASYNCcnt double-buffer pipeline).
// ---------------------------------------------------------------------------

#define WVOCAB 32000
#define EMB    256
#define STEPS  256
#define HID    512
#define QOUT   256
#define BATCH  32
#define LQ     50
#define G4H    2048   // 4*HID gate width

typedef __attribute__((ext_vector_type(16))) __bf16 v16bf;
typedef __attribute__((ext_vector_type(8)))  __bf16 v8bf;
typedef __attribute__((ext_vector_type(8)))  float  v8f;

// A-operand fragment (16-bit A, 16x32 tile): lane L holds row (L&15),
// K = (L>>4)*8 + {0..7} in regs 0..3 and +16 in regs 4..7.
__device__ __forceinline__ v16bf load_a16(const __bf16* p) {
    v8bf lo = *(const v8bf*)(p);
    v8bf hi = *(const v8bf*)(p + 16);
    v16bf r;
#pragma unroll
    for (int i = 0; i < 8; ++i) { r[i] = lo[i]; r[i + 8] = hi[i]; }
    return r;
}

__device__ __forceinline__ float sigm_(float x) { return 1.0f / (1.0f + __expf(-x)); }
__device__ __forceinline__ float tanh_(float x) {
    float e = __expf(2.0f * x);
    return (e - 1.0f) / (e + 1.0f);
}

// ---------------------------------------------------------------------------
// Elementwise helpers
// ---------------------------------------------------------------------------
__global__ void k_cvt(const float* __restrict__ s, __bf16* __restrict__ d, int n) {
    for (int i = blockIdx.x * blockDim.x + threadIdx.x; i < n; i += gridDim.x * blockDim.x)
        d[i] = (__bf16)s[i];
}

__global__ void k_bias2(const float* __restrict__ a, const float* __restrict__ b,
                        float* __restrict__ d, int n) {
    int i = blockIdx.x * blockDim.x + threadIdx.x;
    if (i < n) d[i] = a[i] + b[i];
}

// Gather question embeddings -> bf16 [B*LQ, EMB]
__global__ void k_embed_q(const int* __restrict__ q, const float* __restrict__ emb,
                          __bf16* __restrict__ out) {
    int i = blockIdx.x * blockDim.x + threadIdx.x;   // B*LQ*EMB threads
    int e   = i & (EMB - 1);
    int row = i >> 8;                                 // b*LQ + t
    int tok = q[row];
    out[i] = (__bf16)emb[(size_t)tok * EMB + e];
}

// dec_in[b*STEPS+t] = [ a_emb_w[answer[b,t]] (256) ; q_out[b] (256) ] -> bf16
__global__ void k_embed_dec(const int* __restrict__ ans, const float* __restrict__ emb,
                            const float* __restrict__ qout, __bf16* __restrict__ out) {
    int i = blockIdx.x * blockDim.x + threadIdx.x;   // B*STEPS*512 threads
    int col = i & 511;
    int row = i >> 9;                                 // b*STEPS + t
    int b = row >> 8;
    int t = row & (STEPS - 1);
    float v;
    if (col < EMB) {
        int tok = ans[b * (STEPS + 1) + t];
        v = emb[(size_t)tok * EMB + col];
    } else {
        v = qout[b * QOUT + (col - EMB)];
    }
    out[i] = (__bf16)v;
}

// ---------------------------------------------------------------------------
// Small WMMA GEMM (used for q_out only, M=32): C[M,N] = A*B^T + bias.
// ---------------------------------------------------------------------------
__global__ __launch_bounds__(256) void k_gemm(
    const __bf16* __restrict__ A, const __bf16* __restrict__ B,
    const float* __restrict__ bias, float* __restrict__ C,
    int M, int N, int K) {
    int w    = threadIdx.x >> 5;
    int lane = threadIdx.x & 31;
    int lm   = lane & 15;
    int lk   = lane >> 4;
    int mt   = blockIdx.y;
    int n    = blockIdx.x * 128 + w * 16 + lm;

    const __bf16* arow = A + (size_t)(mt * 16 + lm) * K + lk * 8;
    const __bf16* brow = B + (size_t)n * K + lk * 8;

    v8f acc = {};
#pragma unroll 4
    for (int k = 0; k < K; k += 32) {
        v16bf af = load_a16(arow + k);
        v16bf bf = load_a16(brow + k);
        acc = __builtin_amdgcn_wmma_f32_16x16x32_bf16(
            false, af, false, bf, (short)0, acc, false, false);
    }
    float bv = bias ? bias[n] : 0.0f;
#pragma unroll
    for (int r = 0; r < 8; ++r) {
        int m = mt * 16 + r + lk * 8;
        C[(size_t)m * N + n] = acc[r] + bv;
    }
}

// ---------------------------------------------------------------------------
// Big WMMA GEMM:  C[M,N] = A[M,K] * B[N,K]^T + bias[N]
//   Block tile 64(M) x 128(N); 8 waves, wave w owns 16 cols x 64 rows.
//   A chunk (64x32 bf16 = 4 KB) staged into a 2-deep LDS buffer with
//   global_load_async_to_lds_b128 (one b128 per thread per chunk), software
//   pipelined: issue chunk k+1, s_wait_asynccnt 1, barrier, compute chunk k.
//   out_mode 0: C[m*N+n]  (f32 row-major)
//   out_mode 1: pred layout C[b*(N*STEPS) + n*STEPS + t], m = t*32+b, NT store
// ---------------------------------------------------------------------------
__global__ __launch_bounds__(256) void k_gemm_big(
    const __bf16* __restrict__ A, const __bf16* __restrict__ B,
    const float* __restrict__ bias, float* __restrict__ C,
    int M, int N, int K, int out_mode) {
    __shared__ __bf16 abuf[2][64][40];   // 32-wide chunk + 8 pad, 10.2 KB total

    int tid  = threadIdx.x;
    int w    = tid >> 5;
    int lane = tid & 31;
    int lm   = lane & 15;
    int lk   = lane >> 4;
    int m0   = blockIdx.y * 64;
    int n    = blockIdx.x * 128 + w * 16 + lm;

    // async staging assignment: thread -> (row, 16B segment) of the 64x32 chunk
    int srow = tid >> 2;                 // 0..63
    int sseg = tid & 3;                  // 0..3 (8 bf16 = 16 B)
    const __bf16* gsrc = A + (size_t)(m0 + srow) * K + sseg * 8;
    unsigned ldsdst0 = (unsigned)(size_t)(&abuf[0][srow][sseg * 8]);
    unsigned ldsdst1 = (unsigned)(size_t)(&abuf[1][srow][sseg * 8]);

    const __bf16* brow = B + (size_t)n * K + lk * 8;

    const int KC = K / 32;
    {   // prologue: stage chunk 0 into buffer 0
        unsigned long long ga = (unsigned long long)(size_t)gsrc;
        asm volatile("global_load_async_to_lds_b128 %0, %1, off"
                     :: "v"(ldsdst0), "v"(ga) : "memory");
    }

    v8f acc[4] = {};
    for (int kc = 0; kc < KC; ++kc) {
        int buf = kc & 1;
        if (kc + 1 < KC) {   // stage next chunk into the other buffer
            unsigned long long ga =
                (unsigned long long)(size_t)(gsrc + (size_t)(kc + 1) * 32);
            unsigned dst = ((kc + 1) & 1) ? ldsdst1 : ldsdst0;
            asm volatile("global_load_async_to_lds_b128 %0, %1, off"
                         :: "v"(dst), "v"(ga) : "memory");
            asm volatile("s_wait_asynccnt 1" ::: "memory");   // chunk kc done
        } else {
            asm volatile("s_wait_asynccnt 0" ::: "memory");
        }
        __syncthreads();     // all lanes' async writes for chunk kc visible

        v16bf bf = load_a16(brow + kc * 32);
#pragma unroll
        for (int mt = 0; mt < 4; ++mt) {
            v16bf af = load_a16(&abuf[buf][mt * 16 + lm][lk * 8]);
            acc[mt] = __builtin_amdgcn_wmma_f32_16x16x32_bf16(
                false, af, false, bf, (short)0, acc[mt], false, false);
        }
        __syncthreads();     // reads done before buffer is overwritten
    }

    float bv = bias ? bias[n] : 0.0f;
    if (out_mode == 0) {
#pragma unroll
        for (int mt = 0; mt < 4; ++mt)
#pragma unroll
            for (int r = 0; r < 8; ++r) {
                int m = m0 + mt * 16 + r + lk * 8;
                C[(size_t)m * N + n] = acc[mt][r] + bv;
            }
    } else {
#pragma unroll
        for (int mt = 0; mt < 4; ++mt)
#pragma unroll
            for (int r = 0; r < 8; ++r) {
                int m = m0 + mt * 16 + r + lk * 8;
                int t = m >> 5;
                int b = m & 31;
                __builtin_nontemporal_store(
                    acc[mt][r] + bv,
                    &C[(size_t)b * ((size_t)N * STEPS) + (size_t)n * STEPS + t]);
            }
    }
}

// ---------------------------------------------------------------------------
// Fused LSTM recurrence: one workgroup, 32 waves (1024 threads).
//   gates g = xproj[:, t, :] + h @ whh^T        (h bf16 in LDS, c in regs)
//   wave w owns gate cols [16w,16w+16) of each of the 4 gate blocks, all 32 rows.
//   xproj layout: [(b*T + t) * G4H];  hs_out layout: [(t*32 + b) * HID] (bf16).
// ---------------------------------------------------------------------------
__global__ __launch_bounds__(1024) void k_lstm(
    const float* __restrict__ xproj, const __bf16* __restrict__ whh, int T,
    const __bf16* __restrict__ h0, const float* __restrict__ c0,
    __bf16* __restrict__ hs_out, __bf16* __restrict__ hT_out, float* __restrict__ cT_out) {
    __shared__ __bf16 h_lds[BATCH][HID];   // 32 KB

    int tid  = threadIdx.x;
    int w    = tid >> 5;
    int lane = tid & 31;
    int lm   = lane & 15;
    int lk   = lane >> 4;
    int nb   = w * 16;          // column base within HID
    int nn   = nb + lm;         // this lane's column

    for (int i = tid; i < BATCH * HID; i += 1024) {
        h_lds[i >> 9][i & 511] = h0 ? h0[i] : (__bf16)0.0f;
    }
    float c_reg[2][8];          // lane owns (m = mt*16 + r + 8*lk, n = nn)
#pragma unroll
    for (int mt = 0; mt < 2; ++mt)
#pragma unroll
        for (int r = 0; r < 8; ++r) {
            int m = mt * 16 + r + lk * 8;
            c_reg[mt][r] = c0 ? c0[m * HID + nn] : 0.0f;
        }
    __syncthreads();

    for (int t = 0; t < T; ++t) {
        v8f acc[2][4] = {};   // [m-tile][gate]
#pragma unroll 2
        for (int kc = 0; kc < HID / 32; ++kc) {
            int k = kc * 32;
            v16bf af[2];
#pragma unroll
            for (int mt = 0; mt < 2; ++mt)
                af[mt] = load_a16(&h_lds[mt * 16 + lm][0] + k + lk * 8);
#pragma unroll
            for (int g = 0; g < 4; ++g) {
                const __bf16* bp = whh + (size_t)(g * HID + nb + lm) * HID + k + lk * 8;
                v16bf bf = load_a16(bp);
#pragma unroll
                for (int mt = 0; mt < 2; ++mt)
                    acc[mt][g] = __builtin_amdgcn_wmma_f32_16x16x32_bf16(
                        false, af[mt], false, bf, (short)0, acc[mt][g], false, false);
            }
        }

        float hv[2][8];
#pragma unroll
        for (int mt = 0; mt < 2; ++mt) {
#pragma unroll
            for (int r = 0; r < 8; ++r) {
                int m = mt * 16 + r + lk * 8;
                const float* xp = xproj + ((size_t)m * T + t) * G4H;
                float gi = sigm_(acc[mt][0][r] + xp[nn]);
                float gf = sigm_(acc[mt][1][r] + xp[nn + HID]);
                float gg = tanh_(acc[mt][2][r] + xp[nn + 2 * HID]);
                float go = sigm_(acc[mt][3][r] + xp[nn + 3 * HID]);
                float cn = gf * c_reg[mt][r] + gi * gg;
                c_reg[mt][r] = cn;
                hv[mt][r] = go * tanh_(cn);
            }
        }

        __syncthreads();   // all waves done reading h_lds for this step
#pragma unroll
        for (int mt = 0; mt < 2; ++mt) {
#pragma unroll
            for (int r = 0; r < 8; ++r) {
                int m = mt * 16 + r + lk * 8;
                __bf16 hb = (__bf16)hv[mt][r];
                h_lds[m][nn] = hb;
                if (hs_out) hs_out[(size_t)(t * BATCH + m) * HID + nn] = hb;
            }
        }
        __syncthreads();   // h_lds updated before next step's reads
    }

    if (hT_out) {
        for (int i = tid; i < BATCH * HID; i += 1024)
            hT_out[i] = h_lds[i >> 9][i & 511];
    }
    if (cT_out) {
#pragma unroll
        for (int mt = 0; mt < 2; ++mt)
#pragma unroll
            for (int r = 0; r < 8; ++r) {
                int m = mt * 16 + r + lk * 8;
                cT_out[m * HID + nn] = c_reg[mt][r];
            }
    }
}

// ---------------------------------------------------------------------------
extern "C" void kernel_launch(void* const* d_in, const int* in_sizes, int n_in,
                              void* d_out, int out_size, void* d_ws, size_t ws_size,
                              hipStream_t stream) {
    (void)in_sizes; (void)n_in; (void)out_size; (void)ws_size;
    const int*   question = (const int*)d_in[0];
    const int*   answer   = (const int*)d_in[1];
    const float* q_emb_w  = (const float*)d_in[2];
    const float* a_emb_w  = (const float*)d_in[3];
    const float* q_w_ih   = (const float*)d_in[4];
    const float* q_w_hh   = (const float*)d_in[5];
    const float* q_b_ih   = (const float*)d_in[6];
    const float* q_b_hh   = (const float*)d_in[7];
    const float* q_lin_w  = (const float*)d_in[8];
    const float* q_lin_b  = (const float*)d_in[9];
    const float* a_w_ih   = (const float*)d_in[10];
    const float* a_w_hh   = (const float*)d_in[11];
    const float* a_b_ih   = (const float*)d_in[12];
    const float* a_b_hh   = (const float*)d_in[13];
    const float* lin_w    = (const float*)d_in[14];
    const float* lin_b    = (const float*)d_in[15];
    float* out = (float*)d_out;

    char* p = (char*)d_ws;
    auto alloc = [&](size_t bytes) -> void* {
        void* r = (void*)p;
        p += (bytes + 255) & ~(size_t)255;
        return r;
    };
    __bf16* linw_bf  = (__bf16*)alloc((size_t)WVOCAB * HID * 2);
    __bf16* awih_bf  = (__bf16*)alloc((size_t)G4H * HID * 2);
    __bf16* awhh_bf  = (__bf16*)alloc((size_t)G4H * HID * 2);
    __bf16* qwih_bf  = (__bf16*)alloc((size_t)G4H * EMB * 2);
    __bf16* qwhh_bf  = (__bf16*)alloc((size_t)G4H * HID * 2);
    __bf16* qlinw_bf = (__bf16*)alloc((size_t)QOUT * HID * 2);
    float*  biasq    = (float*)alloc(G4H * 4);
    float*  biasa    = (float*)alloc(G4H * 4);
    __bf16* qe_bf    = (__bf16*)alloc((size_t)BATCH * LQ * EMB * 2);
    float*  xprojq   = (float*)alloc((size_t)BATCH * LQ * G4H * 4);
    __bf16* ench_bf  = (__bf16*)alloc((size_t)BATCH * HID * 2);
    float*  encc     = (float*)alloc((size_t)BATCH * HID * 4);
    float*  qoutf    = (float*)alloc((size_t)BATCH * QOUT * 4);
    __bf16* decin_bf = (__bf16*)alloc((size_t)BATCH * STEPS * HID * 2);
    float*  xprojd   = (float*)alloc((size_t)BATCH * STEPS * G4H * 4);
    __bf16* hs_bf    = (__bf16*)alloc((size_t)STEPS * BATCH * HID * 2);

    // 1) weight conversions f32 -> bf16
    k_cvt<<<2048, 256, 0, stream>>>(lin_w,   linw_bf,  WVOCAB * HID);
    k_cvt<<< 512, 256, 0, stream>>>(a_w_ih,  awih_bf,  G4H * HID);
    k_cvt<<< 512, 256, 0, stream>>>(a_w_hh,  awhh_bf,  G4H * HID);
    k_cvt<<< 512, 256, 0, stream>>>(q_w_ih,  qwih_bf,  G4H * EMB);
    k_cvt<<< 512, 256, 0, stream>>>(q_w_hh,  qwhh_bf,  G4H * HID);
    k_cvt<<< 256, 256, 0, stream>>>(q_lin_w, qlinw_bf, QOUT * HID);
    k_bias2<<<G4H / 256, 256, 0, stream>>>(q_b_ih, q_b_hh, biasq, G4H);
    k_bias2<<<G4H / 256, 256, 0, stream>>>(a_b_ih, a_b_hh, biasa, G4H);

    // 2) encoder: embed -> x-proj GEMM -> recurrence -> q_out
    k_embed_q<<<(BATCH * LQ * EMB) / 256, 256, 0, stream>>>(question, q_emb_w, qe_bf);
    {
        dim3 g(G4H / 128, (BATCH * LQ) / 64);   // M=1600 = 25*64
        k_gemm_big<<<g, 256, 0, stream>>>(qe_bf, qwih_bf, biasq, xprojq,
                                          BATCH * LQ, G4H, EMB, 0);
    }
    k_lstm<<<1, 1024, 0, stream>>>(xprojq, qwhh_bf, LQ,
                                   nullptr, nullptr, nullptr, ench_bf, encc);
    {
        dim3 g(QOUT / 128, BATCH / 16);
        k_gemm<<<g, 256, 0, stream>>>(ench_bf, qlinw_bf, q_lin_b, qoutf,
                                      BATCH, QOUT, HID);
    }

    // 3) decoder: embed+concat -> x-proj GEMM -> recurrence
    k_embed_dec<<<(BATCH * STEPS * HID) / 256, 256, 0, stream>>>(answer, a_emb_w,
                                                                 qoutf, decin_bf);
    {
        dim3 g(G4H / 128, (BATCH * STEPS) / 64);
        k_gemm_big<<<g, 256, 0, stream>>>(decin_bf, awih_bf, biasa, xprojd,
                                          BATCH * STEPS, G4H, HID, 0);
    }
    k_lstm<<<1, 1024, 0, stream>>>(xprojd, awhh_bf, STEPS,
                                   ench_bf, encc, hs_bf, nullptr, nullptr);

    // 4) output projection (dominant GEMM): logits -> pred[B, W, T]
    {
        dim3 g(WVOCAB / 128, (STEPS * BATCH) / 64);
        k_gemm_big<<<g, 256, 0, stream>>>(hs_bf, linw_bf, lin_b, out,
                                          STEPS * BATCH, WVOCAB, HID, 1);
    }
}
